// RelPosAttention_81776177316139
// MI455X (gfx1250) — compile-verified, matrix operand order checked by reading, not run on hardware
//
#include <hip/hip_runtime.h>

#define BB 16
#define NN 1024
#define DD 512
#define HH 16
#define HD 32
#define QK_SCALE 0.17677669529663687f  // 32^-0.5

typedef __attribute__((ext_vector_type(16))) __bf16 bf16x16;
typedef __attribute__((ext_vector_type(8)))  __bf16 bf16x8;
typedef __attribute__((ext_vector_type(4)))  __bf16 bf16x4;
typedef __attribute__((ext_vector_type(8)))  float  f32x8;
typedef __attribute__((ext_vector_type(4)))  float  f32x4;

union Frag16 { bf16x16 v; bf16x8 h[2]; __bf16 e[16]; };

// Fragment loader for both A (lane=row, K striped over VGPRs) and B
// (lane=col, K striped identically). Requires K contiguous in memory with
// leading dimension ldk. Matches the 16-bit 16x32 layout table:
//   lanes 0-15 : K = k0+0..7  (VGPR0..3), k0+16..23 (VGPR4..7)
//   lanes 16-31: K = k0+8..15,            k0+24..31
__device__ __forceinline__ bf16x16 load_frag(const __bf16* __restrict__ base,
                                             int row, int ldk, int k0) {
  const int lane = threadIdx.x & 31;
  const int r    = lane & 15;
  const int hi   = (lane >> 4) & 1;
  const __bf16* p = base + (size_t)(row + r) * (size_t)ldk + k0 + hi * 8;
  Frag16 f;
  f.h[0] = *(const bf16x8*)(p);        // K k0+hi*8 .. +7
  f.h[1] = *(const bf16x8*)(p + 16);   // K k0+16+hi*8 .. +7
  return f.v;
}

__device__ __forceinline__ f32x8 wmma_bf16(bf16x16 a, bf16x16 b, f32x8 c) {
  return __builtin_amdgcn_wmma_f32_16x16x32_bf16(false, a, false, b,
                                                 (short)0, c, false, false);
}

// ---------------- conversion / preprocessing ----------------

__global__ void k_cvt_x(const float* __restrict__ x, __bf16* __restrict__ xb, int n4) {
  int t = blockIdx.x * blockDim.x + threadIdx.x;
  if (t >= n4) return;
  f32x4 v = ((const f32x4*)x)[t];
  bf16x4 o;
  o.x = (__bf16)v.x; o.y = (__bf16)v.y; o.z = (__bf16)v.z; o.w = (__bf16)v.w;
  ((bf16x4*)xb)[t] = o;
}

// w[K][Ncols] fp32 -> wT[Ncols][K] bf16
__global__ void k_tr(const float* __restrict__ w, __bf16* __restrict__ wT,
                     int K, int Ncols) {
  int t = blockIdx.x * blockDim.x + threadIdx.x;
  if (t >= K * Ncols) return;
  int k = t / Ncols;
  int n = t - k * Ncols;
  wT[(size_t)n * K + k] = (__bf16)w[t];
}

// bias_t[h][key][query] = bias_table[rel_index[query][key]][h]   (bf16)
__global__ void k_bias(const float* __restrict__ table, const int* __restrict__ rel,
                       __bf16* __restrict__ biasT) {
  int t = blockIdx.x * blockDim.x + threadIdx.x;   // 0 .. N*N-1
  int n = t & (NN - 1);    // query (fast dim -> coalesced writes)
  int m = t >> 10;         // key
  int idx = rel[n * NN + m];
  const float* trow = table + (size_t)idx * HH;
#pragma unroll
  for (int h = 0; h < HH; ++h)
    biasT[((size_t)h * NN + m) * NN + n] = (__bf16)trow[h];
}

// ---------------- GEMM 1: qkv = x @ w_qkv + b, scatter to Q/K/V ----------------
// 256 thr = 8 waves; wave tile 32x32; WG tile 64(M) x 128(N). M=16384, N=1536, K=512.

__global__ void k_gemm_qkv(const __bf16* __restrict__ xb, const __bf16* __restrict__ wT,
                           const float* __restrict__ bqkv,
                           __bf16* __restrict__ qw, __bf16* __restrict__ kw,
                           __bf16* __restrict__ vw) {
  const int wid  = threadIdx.x >> 5;
  const int lane = threadIdx.x & 31;
  const int wm = wid & 1, wn = wid >> 1;
  const int m0 = blockIdx.x * 64 + wm * 32;
  const int n0 = blockIdx.y * 128 + wn * 32;
  f32x8 acc[2][2] = {};
  for (int k0 = 0; k0 < DD; k0 += 32) {
    bf16x16 a0 = load_frag(xb, m0,      DD, k0);
    bf16x16 a1 = load_frag(xb, m0 + 16, DD, k0);
    bf16x16 b0 = load_frag(wT, n0,      DD, k0);
    bf16x16 b1 = load_frag(wT, n0 + 16, DD, k0);
    acc[0][0] = wmma_bf16(a0, b0, acc[0][0]);
    acc[0][1] = wmma_bf16(a0, b1, acc[0][1]);
    acc[1][0] = wmma_bf16(a1, b0, acc[1][0]);
    acc[1][1] = wmma_bf16(a1, b1, acc[1][1]);
  }
  const int r = lane & 15, hi = (lane >> 4) & 1;
#pragma unroll
  for (int i = 0; i < 2; ++i)
#pragma unroll
    for (int j = 0; j < 2; ++j) {
      int c = n0 + j * 16 + r;
      int which = c >> 9;            // 0=q 1=k 2=v (uniform per 32-wide tile)
      int hh = (c >> 5) & 15;        // head (uniform per tile)
      int d  = c & 31;
      float bias = bqkv[c];
#pragma unroll
      for (int v = 0; v < 8; ++v) {
        int m = m0 + i * 16 + v + hi * 8;
        int b = m >> 10, tok = m & (NN - 1);
        float val = acc[i][j][v] + bias;
        if (which == 0)
          qw[((size_t)((b * HH + hh) * NN + tok)) * HD + d] = (__bf16)(val * QK_SCALE);
        else if (which == 1)
          kw[((size_t)((b * HH + hh) * NN + tok)) * HD + d] = (__bf16)val;
        else
          vw[((size_t)((b * HH + hh) * HD + d)) * NN + tok] = (__bf16)val;
      }
    }
}

// ---------------- flash attention, one wave per (b, h, 16-query block) ----------------
// Computes T = K_block x Q^T (keys over VGPRs, queries over lanes); T's C-layout
// doubles as the A-fragment of P for O += P x V (no transpose needed).

__global__ void k_attn(const __bf16* __restrict__ qw, const __bf16* __restrict__ kw,
                       const __bf16* __restrict__ vw, const __bf16* __restrict__ biasT,
                       __bf16* __restrict__ ow) {
  const int wid  = threadIdx.x >> 5;
  const int lane = threadIdx.x & 31;
  int wg = blockIdx.x * 8 + wid;
  int qb = wg & 63, h = (wg >> 6) & 15, b = wg >> 10;
  int q0 = qb << 4;

  const __bf16* qbase = qw + (size_t)(b * HH + h) * NN * HD;
  const __bf16* kbase = kw + (size_t)(b * HH + h) * NN * HD;
  const __bf16* vbase = vw + (size_t)(b * HH + h) * HD * NN;
  const __bf16* bbase = biasT + (size_t)h * NN * NN;

  bf16x16 qf = load_frag(qbase, q0, HD, 0);   // B-frag: Q^T (fixed over key loop)
  const int r = lane & 15, hi = (lane >> 4) & 1;
  const int permbase = (lane & 16) ? 96 : 0;  // row v -> lane v (lo) / v+24 (hi)

  float mrun = -1e30f, lrun = 0.f;
  f32x8 o0 = {}, o1 = {};

  for (int key0 = 0; key0 < NN; key0 += 32) {
    f32x8 z = {};
    bf16x16 ak0 = load_frag(kbase, key0,      HD, 0);
    bf16x16 ak1 = load_frag(kbase, key0 + 16, HD, 0);
    f32x8 t0 = wmma_bf16(ak0, qf, z);   // keys key0+v(+8)   x queries (lanes)
    f32x8 t1 = wmma_bf16(ak1, qf, z);   // keys key0+16+v(+8)
#pragma unroll
    for (int v = 0; v < 8; ++v) {       // + relative position bias
      int kr = key0 + v + hi * 8;
      t0[v] += (float)bbase[(size_t)kr * NN + q0 + r];
      t1[v] += (float)bbase[(size_t)(kr + 16) * NN + q0 + r];
    }
    // online softmax over keys: per-lane reduction + lane-pair combine
    float mb = t0[0];
#pragma unroll
    for (int v = 0; v < 8; ++v) { mb = fmaxf(mb, t0[v]); mb = fmaxf(mb, t1[v]); }
    mb = fmaxf(mb, __shfl_xor(mb, 16));
    float mnew  = fmaxf(mrun, mb);
    float alpha = __expf(mrun - mnew);
    float s = 0.f;
#pragma unroll
    for (int v = 0; v < 8; ++v) {
      float p0 = __expf(t0[v] - mnew);
      float p1 = __expf(t1[v] - mnew);
      t0[v] = p0; t1[v] = p1; s += p0 + p1;
    }
    s += __shfl_xor(s, 16);
    lrun = lrun * alpha + s;
    mrun = mnew;
    // rescale O rows (row index lives over VGPRs -> broadcast alpha per row)
#pragma unroll
    for (int v = 0; v < 8; ++v) {
      float av = __int_as_float(
          __builtin_amdgcn_ds_bpermute(permbase + v * 4, __float_as_int(alpha)));
      o0[v] *= av; o1[v] *= av;
    }
    // P (C-layout of T) == A-fragment: pure per-lane bf16 packing
    Frag16 pa;
#pragma unroll
    for (int v = 0; v < 8; ++v) { pa.e[v] = (__bf16)t0[v]; pa.e[8 + v] = (__bf16)t1[v]; }
    bf16x16 bv0 = load_frag(vbase, 0,  NN, key0);   // V cols d=0..15, K=keys
    bf16x16 bv1 = load_frag(vbase, 16, NN, key0);   // V cols d=16..31
    o0 = wmma_bf16(pa.v, bv0, o0);
    o1 = wmma_bf16(pa.v, bv1, o1);
  }

  float linv = 1.f / lrun;
#pragma unroll
  for (int v = 0; v < 8; ++v) {
    float lv = __int_as_float(
        __builtin_amdgcn_ds_bpermute(permbase + v * 4, __float_as_int(linv)));
    int tok = q0 + v + hi * 8;
    size_t base = ((size_t)(b * NN + tok)) * DD + h * HD;
    ow[base + r]      = (__bf16)(o0[v] * lv);
    ow[base + 16 + r] = (__bf16)(o1[v] * lv);
  }
}

// ---------------- GEMM 2: out = O @ w_proj + b_proj (fp32 out) ----------------

__global__ void k_gemm_proj(const __bf16* __restrict__ ob, const __bf16* __restrict__ wT,
                            const float* __restrict__ bp, float* __restrict__ out) {
  const int wid  = threadIdx.x >> 5;
  const int lane = threadIdx.x & 31;
  const int wm = wid & 1, wn = wid >> 1;
  const int m0 = blockIdx.x * 64 + wm * 32;
  const int n0 = blockIdx.y * 128 + wn * 32;
  f32x8 acc[2][2] = {};
  for (int k0 = 0; k0 < DD; k0 += 32) {
    bf16x16 a0 = load_frag(ob, m0,      DD, k0);
    bf16x16 a1 = load_frag(ob, m0 + 16, DD, k0);
    bf16x16 b0 = load_frag(wT, n0,      DD, k0);
    bf16x16 b1 = load_frag(wT, n0 + 16, DD, k0);
    acc[0][0] = wmma_bf16(a0, b0, acc[0][0]);
    acc[0][1] = wmma_bf16(a0, b1, acc[0][1]);
    acc[1][0] = wmma_bf16(a1, b0, acc[1][0]);
    acc[1][1] = wmma_bf16(a1, b1, acc[1][1]);
  }
  const int r = lane & 15, hi = (lane >> 4) & 1;
#pragma unroll
  for (int i = 0; i < 2; ++i)
#pragma unroll
    for (int j = 0; j < 2; ++j) {
      int c = n0 + j * 16 + r;
      float bias = bp[c];
#pragma unroll
      for (int v = 0; v < 8; ++v) {
        int m = m0 + i * 16 + v + hi * 8;
        out[(size_t)m * DD + c] = acc[i][j][v] + bias;
      }
    }
}

// ---------------- host ----------------

extern "C" void kernel_launch(void* const* d_in, const int* in_sizes, int n_in,
                              void* d_out, int out_size, void* d_ws, size_t ws_size,
                              hipStream_t stream) {
  (void)in_sizes; (void)n_in; (void)out_size; (void)ws_size;
  const float* x      = (const float*)d_in[0];
  const float* w_qkv  = (const float*)d_in[1];
  const float* b_qkv  = (const float*)d_in[2];
  const float* w_proj = (const float*)d_in[3];
  const float* b_proj = (const float*)d_in[4];
  const float* table  = (const float*)d_in[5];
  const int*   rel    = (const int*)d_in[6];
  float* out = (float*)d_out;

  char* ws = (char*)d_ws;
  // workspace map (bytes)
  __bf16* xb  = (__bf16*)(ws);                 // x bf16           16.78 MB
  __bf16* wqT = (__bf16*)(ws + 16777216);      // w_qkv^T bf16      1.57 MB
  __bf16* wpT = (__bf16*)(ws + 18350080);      // w_proj^T bf16     0.52 MB
  __bf16* qw  = (__bf16*)(ws + 18874368);      // Q [b,h,n,d]      16.78 MB
  __bf16* kw  = (__bf16*)(ws + 35651584);      // K [b,h,n,d]      16.78 MB
  __bf16* vw  = (__bf16*)(ws + 52428800);      // V [b,h,d,n]      16.78 MB
  __bf16* ow  = (__bf16*)(ws + 69206016);      // O [b,n,512]      16.78 MB
  __bf16* bT  = (__bf16*)(ws + 85983232);      // bias [h,key,q]   33.55 MB

  k_cvt_x    <<<dim3(8192),    dim3(256), 0, stream>>>(x, xb, (BB * NN * DD) / 4);
  k_tr       <<<dim3(3072),    dim3(256), 0, stream>>>(w_qkv, wqT, DD, 3 * DD);
  k_tr       <<<dim3(1024),    dim3(256), 0, stream>>>(w_proj, wpT, DD, DD);
  k_bias     <<<dim3(4096),    dim3(256), 0, stream>>>(table, rel, bT);
  k_gemm_qkv <<<dim3(256, 12), dim3(256), 0, stream>>>(xb, wqT, b_qkv, qw, kw, vw);
  k_attn     <<<dim3(2048),    dim3(256), 0, stream>>>(qw, kw, vw, bT, ow);
  k_gemm_proj<<<dim3(256, 4),  dim3(256), 0, stream>>>(ow, wpT, b_proj, out);
}
